// Backflow_11227044512058
// MI455X (gfx1250) — compile-verified, matrix operand order checked by reading, not run on hardware
//
#include <hip/hip_runtime.h>
#include <hip/hip_bf16.h>
#include <math.h>

// ---------------------------------------------------------------------------
// Backflow for MI455X (gfx1250, wave32, WMMA).
//
// Inputs (float32, setup_inputs order):
//  0 rs    [128,64,3]     1 xs  [128,64,128]   2 coords [8,3]
//  3 We1 [128,25] 4 be1[25] 5 We2[25,5] 6 be2[5] 7 We3[5,1] 8 be3[1]
//  9 Wn1 [128,51] 10 bn1[51] 11 Wn2[51,20] 12 bn2[20] 13 Wn3[20,8] 14 bn3[8]
// Output: [128,64,3] float32.
//
// Kernel 1 (VALU): nuclear MLP + cutoff per (b,n) -> ws[rid*4] = {bf_nuc, scale}
// Kernel 2 (WMMA): electron-pair MLP layer1 via v_wmma_f32_16x16x32_f16,
//                  layers 2/3 on VALU, fused final combine.
//
// ssp() uses hardware transcendentals (v_exp_f32 / v_log_f32): the backflow
// term is scaled by 1e-4 before being added to O(1) coordinates and the WMMA
// inputs are f16 anyway, so ~1e-6 relative error in softplus is invisible.
// ---------------------------------------------------------------------------

typedef _Float16 v16h __attribute__((ext_vector_type(16)));
typedef _Float16 v8h  __attribute__((ext_vector_type(8)));
typedef float    v8f  __attribute__((ext_vector_type(8)));

#define LN2F 0.6931471805599453f

__device__ __forceinline__ float sspf(float x) {
    // shifted softplus: max(x,0) + log(1 + exp(-|x|)) - ln2
    // arg of log is in (1,2], so fast __logf is well-conditioned here.
    return fmaxf(x, 0.0f) + __logf(1.0f + __expf(-fabsf(x))) - LN2F;
}

// ---------------------------------------------------------------------------
// Kernel 1: nuclear backflow + cutoff. One wave per (b,n) row, 8 rows/block.
// ---------------------------------------------------------------------------
__global__ __launch_bounds__(256)
void nuc_kernel(const float* __restrict__ rs, const float* __restrict__ xs,
                const float* __restrict__ coords,
                const float* __restrict__ Wn1, const float* __restrict__ bn1,
                const float* __restrict__ Wn2, const float* __restrict__ bn2,
                const float* __restrict__ Wn3, const float* __restrict__ bn3,
                float* __restrict__ ws) {
    __shared__ float xb[8][128];
    __shared__ float h1[8][52];
    __shared__ float h2[8][20];

    const int tid  = threadIdx.x;
    const int lane = tid & 31;
    const int wave = tid >> 5;
    const int rid  = blockIdx.x * 8 + wave;       // 0 .. 8191

    // stage x row into LDS (4 floats per lane)
    const float* xr = xs + (size_t)rid * 128;
    for (int t = 0; t < 4; ++t) xb[wave][lane * 4 + t] = xr[lane * 4 + t];
    __syncthreads();

    // layer 1: 128 -> 51 (lane-parallel over outputs)
    for (int o = lane; o < 51; o += 32) {
        float acc = bn1[o];
        #pragma unroll 8
        for (int k = 0; k < 128; ++k) acc = fmaf(xb[wave][k], Wn1[k * 51 + o], acc);
        h1[wave][o] = sspf(acc);
    }
    __syncthreads();

    // layer 2: 51 -> 20
    if (lane < 20) {
        float acc = bn2[lane];
        for (int k = 0; k < 51; ++k) acc = fmaf(h1[wave][k], Wn2[k * 20 + lane], acc);
        h2[wave][lane] = sspf(acc);
    }
    __syncthreads();

    // layer 3: 20 -> 8, plus per-nucleus diff / cutoff (lanes 0..7 = nuclei)
    float vx = 0.f, vy = 0.f, vz = 0.f, cut = 1.0f;
    if (lane < 8) {
        float g = bn3[lane];
        for (int k = 0; k < 20; ++k) g = fmaf(h2[wave][k], Wn3[k * 8 + lane], g);
        const float px = rs[rid * 3 + 0], py = rs[rid * 3 + 1], pz = rs[rid * 3 + 2];
        const float dx = px - coords[lane * 3 + 0];
        const float dy = py - coords[lane * 3 + 1];
        const float dz = pz - coords[lane * 3 + 2];
        const float r  = sqrtf(dx * dx + dy * dy + dz * dz);
        // reference: t = r/L (L=0.5); where(t < L): t^2*(6-8t+3t^2) else 1
        const float t = 2.0f * r;
        cut = (t < 0.5f) ? t * t * (6.0f - 8.0f * t + 3.0f * t * t) : 1.0f;
        vx = g * dx; vy = g * dy; vz = g * dz;
    }
    // reduce lanes 0..7: sum vec, product cutoff
    for (int off = 4; off; off >>= 1) {
        vx  += __shfl_xor(vx,  off, 8);
        vy  += __shfl_xor(vy,  off, 8);
        vz  += __shfl_xor(vz,  off, 8);
        cut *= __shfl_xor(cut, off, 8);
    }
    if (lane == 0) {
        float* o = ws + (size_t)rid * 4;
        o[0] = vx; o[1] = vy; o[2] = vz; o[3] = 1e-4f * cut;
    }
}

// ---------------------------------------------------------------------------
// Kernel 2: electronic backflow via WMMA. One workgroup per batch element.
// Pair grid is the full 64x64 (diagonal self-cancels via rs_i - rs_i = 0).
// Wave w owns i in [8w, 8w+8); per i: 4 j-tiles of 16 -> one 16x16x128 GEMM
// against We1 (25 cols padded to 32), done as 4 x v_wmma_f32_16x16x32_f16
// per N-tile with accumulators held in VGPRs.
// ---------------------------------------------------------------------------
__global__ __launch_bounds__(256)
void elec_kernel(const float* __restrict__ rs, const float* __restrict__ xs,
                 const float* __restrict__ We1, const float* __restrict__ be1,
                 const float* __restrict__ We2, const float* __restrict__ be2,
                 const float* __restrict__ We3, const float* __restrict__ be3,
                 const float* __restrict__ ws, float* __restrict__ out) {
    __shared__ __align__(16) _Float16 xh[64][128];   // 16 KB: xs[b] in f16
    __shared__ float rsh[64][3];
    __shared__ float be1s[26];
    __shared__ float w2s[125];
    __shared__ float b2s[5];
    __shared__ float w3s[5];
    __shared__ float b3s[1];
    __shared__ float h1buf[8][16][26];               // per-wave transpose buffer

    const int tid  = threadIdx.x;
    const int b    = blockIdx.x;
    const int lane = tid & 31;
    const int wave = tid >> 5;
    const int col  = lane & 15;                      // A-row-in-tile == B/C column
    const int khi  = (lane >> 4) << 3;               // 0 or 8 (K sub-offset)

    // ---- stage batch data into LDS ----
    const float* xsb = xs + (size_t)b * 64 * 128;
    for (int idx = tid; idx < 64 * 128; idx += 256)
        xh[idx >> 7][idx & 127] = (_Float16)xsb[idx];
    if (tid < 64) {
        rsh[tid][0] = rs[((size_t)b * 64 + tid) * 3 + 0];
        rsh[tid][1] = rs[((size_t)b * 64 + tid) * 3 + 1];
        rsh[tid][2] = rs[((size_t)b * 64 + tid) * 3 + 2];
    }
    if (tid < 25)  be1s[tid] = be1[tid];
    if (tid < 125) w2s[tid]  = We2[tid];
    if (tid < 5)   { b2s[tid] = be2[tid]; w3s[tid] = We3[tid]; }
    if (tid == 0)  b3s[0] = be3[0];
    __syncthreads();

    // ---- build We1 B-fragments once, hold in registers (8 x v16h) ----
    // B 32x16 f16 layout: lane holds column n = lane&15; element e maps to
    // k = (e&7) + ((e>>3)<<4) + ((lane>>4)<<3)   (within the 32-K chunk)
    union V16 { v16h v; _Float16 e[16]; };
    V16 bf[4][2];
    for (int kc = 0; kc < 4; ++kc)
        for (int nt = 0; nt < 2; ++nt) {
            const int n = nt * 16 + col;
            for (int e = 0; e < 16; ++e) {
                const int k = kc * 32 + (e & 7) + ((e >> 3) << 4) + khi;
                bf[kc][nt].e[e] = (n < 25) ? (_Float16)We1[k * 25 + n]
                                           : (_Float16)0.0f;
            }
        }

    union V8F { v8f v; float f[8]; };

    for (int iq = 0; iq < 8; ++iq) {
        const int ii = wave * 8 + iq;                // this wave's electron i
        float ax = 0.f, ay = 0.f, az = 0.f;          // per-lane partial bf_elec

        for (int jt = 0; jt < 4; ++jt) {
            const int j = jt * 16 + col;             // this lane's partner row

            // layer 1: h1 = (x_i ⊙ x_j) @ We1  via WMMA, K=128 in 4 chunks
            V8F c0, c1; c0.v = (v8f){}; c1.v = (v8f){};
            #pragma unroll
            for (int kc = 0; kc < 4; ++kc) {
                const int kb = kc * 32 + khi;
                union { v16h v; v8h h[2]; } a;
                const v8h* pi = (const v8h*)&xh[ii][kb];
                const v8h* pj = (const v8h*)&xh[j][kb];
                a.h[0] = pi[0] * pj[0];              // K kb .. kb+7
                a.h[1] = pi[2] * pj[2];              // K kb+16 .. kb+23
                c0.v = __builtin_amdgcn_wmma_f32_16x16x32_f16(
                    false, a.v, false, bf[kc][0].v, (short)0, c0.v, false, false);
                c1.v = __builtin_amdgcn_wmma_f32_16x16x32_f16(
                    false, a.v, false, bf[kc][1].v, (short)0, c1.v, false, false);
            }

            // bias + ssp, transpose C (col-per-lane) -> row-per-lane via LDS
            const int rbase = (lane >> 4) * 8;       // C rows r+8*(lane>>4)
            #pragma unroll
            for (int r = 0; r < 8; ++r) {
                const int row = rbase + r;
                h1buf[wave][row][col] = sspf(c0.f[r] + be1s[col]);
                if (col + 16 < 25)
                    h1buf[wave][row][col + 16] = sspf(c1.f[r] + be1s[col + 16]);
            }
            __syncthreads();

            // layers 2/3 (25->5->1) + weighted diff, one pair-row per lane
            if (lane < 16) {
                float h1r[25];
                #pragma unroll
                for (int k = 0; k < 25; ++k) h1r[k] = h1buf[wave][lane][k];
                float a2[5];
                #pragma unroll
                for (int o = 0; o < 5; ++o) {
                    float acc = b2s[o];
                    #pragma unroll
                    for (int k = 0; k < 25; ++k)
                        acc = fmaf(h1r[k], w2s[k * 5 + o], acc);
                    a2[o] = sspf(acc);
                }
                float h3 = b3s[0];
                #pragma unroll
                for (int o = 0; o < 5; ++o) h3 = fmaf(a2[o], w3s[o], h3);
                const int jr = jt * 16 + lane;       // j==ii gives zero diff
                ax = fmaf(h3, rsh[ii][0] - rsh[jr][0], ax);
                ay = fmaf(h3, rsh[ii][1] - rsh[jr][1], ay);
                az = fmaf(h3, rsh[ii][2] - rsh[jr][2], az);
            }
            __syncthreads();
        }

        // reduce the 16 partner-lanes, combine with nuclear part, write out
        for (int off = 8; off; off >>= 1) {
            ax += __shfl_xor(ax, off, 16);
            ay += __shfl_xor(ay, off, 16);
            az += __shfl_xor(az, off, 16);
        }
        if (lane == 0) {
            const int rid = b * 64 + ii;
            const float* nb = ws + (size_t)rid * 4;
            const float s = nb[3];                   // 1e-4 * cutoff
            out[rid * 3 + 0] = rsh[ii][0] + s * (ax + nb[0]);
            out[rid * 3 + 1] = rsh[ii][1] + s * (ay + nb[1]);
            out[rid * 3 + 2] = rsh[ii][2] + s * (az + nb[2]);
        }
    }
}

// ---------------------------------------------------------------------------
extern "C" void kernel_launch(void* const* d_in, const int* in_sizes, int n_in,
                              void* d_out, int out_size, void* d_ws, size_t ws_size,
                              hipStream_t stream) {
    const float* rs     = (const float*)d_in[0];
    const float* xs     = (const float*)d_in[1];
    const float* coords = (const float*)d_in[2];
    const float* We1 = (const float*)d_in[3];
    const float* be1 = (const float*)d_in[4];
    const float* We2 = (const float*)d_in[5];
    const float* be2 = (const float*)d_in[6];
    const float* We3 = (const float*)d_in[7];
    const float* be3 = (const float*)d_in[8];
    const float* Wn1 = (const float*)d_in[9];
    const float* bn1 = (const float*)d_in[10];
    const float* Wn2 = (const float*)d_in[11];
    const float* bn2 = (const float*)d_in[12];
    const float* Wn3 = (const float*)d_in[13];
    const float* bn3 = (const float*)d_in[14];
    float* out = (float*)d_out;
    float* ws  = (float*)d_ws;      // needs 128*64*4 floats = 128 KB

    nuc_kernel<<<1024, 256, 0, stream>>>(rs, xs, coords,
                                         Wn1, bn1, Wn2, bn2, Wn3, bn3, ws);
    elec_kernel<<<128, 256, 0, stream>>>(rs, xs, We1, be1, We2, be2, We3, be3,
                                         ws, out);
}
